// MemMambaBlock_82214263980184
// MI455X (gfx1250) — matrix operand order repaired and checked by reading, not compiled, not run on hardware
//
#include <hip/hip_runtime.h>
#include <hip/hip_bf16.h>
#include <math.h>

// ---------------- model constants ----------------
#define TLEN   1024
#define BSZ    2
#define BT     (BSZ*TLEN)      // 2048 tokens
#define DMODEL 1024
#define DINNER 2048
#define NH     32
#define HD     64
#define DSTATE 128
#define CONVD  2304            // DINNER + 2*DSTATE
#define DPROJ  4384            // 2*DINNER + 2*DSTATE + NH
#define CHUNKL 256
#define NCHUNK 4
#define ZB     (BSZ*NCHUNK*NH) // 256 (b,c,h) batch for SSD GEMMs
#define SCHID  256
#define SUMD   64

typedef __attribute__((ext_vector_type(16))) __bf16 bf16x16;
typedef __attribute__((ext_vector_type(8)))  float  f32x8;

__device__ __forceinline__ float dsigm(float x){ return 1.f/(1.f+__expf(-x)); }

// ---------------- generic WMMA GEMM ----------------
// One wave computes a 32x32 fp32 tile (2x2 subtiles of 16x16) of
// C = A(MxK) x B(KxN), K%32==0, N%32==0, via v_wmma_f32_16x16x32_bf16.
// Fragment reuse: each A/B fragment feeds two WMMAs (2x arithmetic intensity).
// Fragment layouts (wave32, 16-bit A 16x32 / B 32x16 / f32 C 16x16): A/B both use k = 16*hi + e; C per ISA.
template <class FA, class FB, class FE>
__global__ __launch_bounds__(128) void gemm_k(FA fa, FB fb, FE fe, int M, int N, int K)
{
    const int lane = threadIdx.x & 31;
    const int wave = threadIdx.x >> 5;
    const int z    = blockIdx.z;
    const int m0   = (blockIdx.y * 4 + wave) * 32;   // 4 waves stacked along M
    const int n0   = blockIdx.x * 32;
    if (m0 >= M) return;
    const int lo = lane & 15, hi = lane >> 4;

    f32x8 acc00 = {0.f,0.f,0.f,0.f,0.f,0.f,0.f,0.f};
    f32x8 acc01 = acc00, acc10 = acc00, acc11 = acc00;

    for (int k0 = 0; k0 < K; k0 += 32) {
        bf16x16 a0, a1, b0, b1;
#pragma unroll
        for (int e = 0; e < 16; ++e) {
            // A: lane lo holds row; K = k0 + hi*16 + e
            int ka = k0 + (hi << 4) + e;
            a0[e] = (__bf16)fa(z, m0 + lo,      ka);
            a1[e] = (__bf16)fa(z, m0 + 16 + lo, ka);
            // B: lane lo holds col; K = k0 + hi*16 + e (coalesced across lanes)
            int kb = k0 + (hi << 4) + e;
            b0[e] = (__bf16)fb(z, kb, n0 + lo);
            b1[e] = (__bf16)fb(z, kb, n0 + 16 + lo);
        }
        acc00 = __builtin_amdgcn_wmma_f32_16x16x32_bf16(false, a0, false, b0,
                                                        (short)0, acc00, false, false);
        acc01 = __builtin_amdgcn_wmma_f32_16x16x32_bf16(false, a0, false, b1,
                                                        (short)0, acc01, false, false);
        acc10 = __builtin_amdgcn_wmma_f32_16x16x32_bf16(false, a1, false, b0,
                                                        (short)0, acc10, false, false);
        acc11 = __builtin_amdgcn_wmma_f32_16x16x32_bf16(false, a1, false, b1,
                                                        (short)0, acc11, false, false);
    }
    // C/D layout: vgpr r, lane -> (M = r + 8*hi, N = lo)
#pragma unroll
    for (int r = 0; r < 8; ++r) {
        int mr = m0 + (hi << 3) + r;
        fe.store(z, mr,      n0 + lo,      acc00[r]);
        fe.store(z, mr,      n0 + 16 + lo, acc01[r]);
        fe.store(z, mr + 16, n0 + lo,      acc10[r]);
        fe.store(z, mr + 16, n0 + 16 + lo, acc11[r]);
    }
}

// ---------------- fragment loaders ----------------
struct LdA {            // A[z][m][k], row-major, optional z-shift for batch sharing
    const float* p; int ld; long bs; int sh;
    __device__ float operator()(int z,int m,int k) const {
        return p[(long)(z >> sh) * bs + (long)m * ld + k];
    }
};
struct LdB {            // B[z][k][n], row-major
    const float* p; int ld; long bs; int sh;
    __device__ float operator()(int z,int k,int n) const {
        return p[(long)(z >> sh) * bs + (long)k * ld + n];
    }
};
struct LdBT {           // B(k,n) = P[z][n][k]  (transposed operand)
    const float* p; int ld; long bs; int sh;
    __device__ float operator()(int z,int k,int n) const {
        return p[(long)(z >> sh) * bs + (long)n * ld + k];
    }
};
// SSD y_diag A: S(l,s) = (s<=l) ? CB[b,c][l][s] * exp(cs[l]-cs[s]) : 0
struct LdSegA {
    const float* CB; const float* cs;
    __device__ float operator()(int z,int l,int s) const {
        if (s > l) return 0.f;
        const float* c = cs + (long)z * CHUNKL;
        return CB[(long)(z >> 5) * (CHUNKL*CHUNKL) + (long)l * CHUNKL + s]
               * __expf(c[l] - c[s]);
    }
};
// SSD y_diag B: xd[token=(z>>5)*256+s][ (z&31)*64 + p ]
struct LdXdB {
    const float* xd;
    __device__ float operator()(int z,int s,int p) const {
        return xd[((long)((z >> 5) * CHUNKL + s)) * DINNER + (z & 31) * HD + p];
    }
};
// SSD states A: A(p,l) = xd[l,h,p] * exp(cs[last]-cs[l])
struct LdDecayA {
    const float* xd; const float* cs;
    __device__ float operator()(int z,int p,int l) const {
        const float* c = cs + (long)z * CHUNKL;
        float dec = __expf(c[CHUNKL-1] - c[l]);
        return xd[((long)((z >> 5) * CHUNKL + l)) * DINNER + (z & 31) * HD + p] * dec;
    }
};

// ---------------- epilogues ----------------
struct EpStore {
    float* c; int ld; long bs;
    __device__ void store(int z,int m,int n,float v) const {
        c[(long)z * bs + (long)m * ld + n] = v;
    }
};
struct EpRelu {
    float* c; int ld;
    __device__ void store(int z,int m,int n,float v) const {
        c[(long)m * ld + n] = fmaxf(v, 0.f);
    }
};
struct EpAddSig {       // c = sigmoid(c + v)   (split-K gate GEMM pass 2)
    float* c; int ld;
    __device__ void store(int z,int m,int n,float v) const {
        long i = (long)m * ld + n;
        c[i] = dsigm(c[i] + v);
    }
};
struct EpYdiag {        // y_ssd[token][h*64+p] = v
    float* y;
    __device__ void store(int z,int l,int p,float v) const {
        y[((long)((z >> 5) * CHUNKL + l)) * DINNER + (z & 31) * HD + p] = v;
    }
};
struct EpYoff {         // y_ssd += exp(cs[l])*v + D[h]*xs[l,h,p]
    float* y; const float* cs; const float* xs; const float* Dv;
    __device__ void store(int z,int l,int p,float v) const {
        int h = z & 31; long tok = (long)(z >> 5) * CHUNKL + l;
        long yi = tok * DINNER + h * HD + p;
        y[yi] += __expf(cs[(long)z * CHUNKL + l]) * v
                 + Dv[h] * xs[tok * CONVD + h * HD + p];
    }
};

// ---------------- elementwise / scan kernels ----------------
__global__ void rmsnorm1_k(const float* x, const float* w, float* o) {
    __shared__ float red[256];
    long row = blockIdx.x; int tid = threadIdx.x;
    const float* xr = x + row * DMODEL;
    float s = 0.f;
    for (int j = tid; j < DMODEL; j += 256) { float v = xr[j]; s += v * v; }
    red[tid] = s; __syncthreads();
    for (int st = 128; st > 0; st >>= 1) {
        if (tid < st) red[tid] += red[tid + st];
        __syncthreads();
    }
    float inv = rsqrtf(red[0] / DMODEL + 1e-5f);
    for (int j = tid; j < DMODEL; j += 256) o[row * DMODEL + j] = xr[j] * inv * w[j];
}

__global__ void conv_k(const float* zx, const float* cw, const float* cb, float* xBC) {
    long i = (long)blockIdx.x * 256 + threadIdx.x;
    if (i >= (long)BT * CONVD) return;
    int ch = (int)(i % CONVD); long bt = i / CONVD;
    int t = (int)(bt & (TLEN - 1)); int b = (int)(bt >> 10);
    float acc = cb[ch];
#pragma unroll
    for (int k = 0; k < 4; ++k) {
        int ts = t + k - 3;
        if (ts >= 0)
            acc += zx[((long)(b * TLEN + ts)) * DPROJ + DINNER + ch] * cw[ch * 4 + k];
    }
    xBC[i] = acc * dsigm(acc);   // SiLU
}

__global__ void dt_k(const float* zx, const float* dt_bias, const float* A_log,
                     const float* xBC, float* xd, float* dA) {
    int i = blockIdx.x * blockDim.x + threadIdx.x;
    if (i >= BT * NH) return;
    int h = i & 31; long bt = i >> 5;
    float raw = zx[bt * DPROJ + (DPROJ - NH) + h] + dt_bias[h];
    float dt = (raw > 20.f) ? raw : log1pf(__expf(raw));
    dA[i] = dt * (-__expf(A_log[h]));
    for (int p = 0; p < HD; ++p)
        xd[bt * DINNER + h * HD + p] = xBC[bt * CONVD + h * HD + p] * dt;
}

__global__ void cumsum_k(const float* dA, float* cs) {
    int h = threadIdx.x; int bc = blockIdx.x;          // 8 blocks x 32 threads
    float run = 0.f;
    for (int l = 0; l < CHUNKL; ++l) {
        run += dA[((long)bc * CHUNKL + l) * NH + h];
        cs[((long)bc * NH + h) * CHUNKL + l] = run;
    }
}

__global__ void prev_k(const float* states, const float* cs, float* prev) {
    int i = blockIdx.x * blockDim.x + threadIdx.x;     // 2*4*32*64*128
    if (i >= BSZ * NCHUNK * NH * HD * DSTATE) return;
    int n = i & 127, p = (i >> 7) & 63, h = (i >> 13) & 31;
    int zc = (i >> 18) & 3, b = i >> 20;
    float acc = 0.f;
    for (int j = 0; j < zc; ++j) {
        float seg = 0.f;
        for (int c2 = j + 1; c2 <= zc - 1; ++c2)
            seg += cs[((long)((b * NCHUNK + c2) * NH + h)) * CHUNKL + (CHUNKL - 1)];
        acc += __expf(seg) *
               states[((long)((b * NCHUNK + j) * NH + h)) * (HD*DSTATE) + p * DSTATE + n];
    }
    prev[((long)((b * NCHUNK + zc) * NH + h)) * (HD*DSTATE) + p * DSTATE + n] = acc;
}

__global__ void gatednorm_k(const float* y, const float* zx, const float* w, float* o) {
    __shared__ float red[256];
    long row = blockIdx.x; int tid = threadIdx.x;
    const float* yr = y + row * DINNER;
    const float* zr = zx + row * DPROJ;                 // z = first DINNER cols
    float s = 0.f;
    for (int j = tid; j < DINNER; j += 256) {
        float zv = zr[j]; float t = yr[j] * zv * dsigm(zv);
        s += t * t;
    }
    red[tid] = s; __syncthreads();
    for (int st = 128; st > 0; st >>= 1) {
        if (tid < st) red[tid] += red[tid + st];
        __syncthreads();
    }
    float inv = rsqrtf(red[0] / DINNER + 1e-5f);
    for (int j = tid; j < DINNER; j += 256) {
        float zv = zr[j]; float t = yr[j] * zv * dsigm(zv);
        o[row * DINNER + j] = t * inv * w[j];
    }
}

__global__ void scorer2_k(const float* hid, const float* w2, float* scores) {
    __shared__ float red[256];
    long row = blockIdx.x; int tid = threadIdx.x;
    red[tid] = hid[row * SCHID + tid] * w2[tid];
    __syncthreads();
    for (int st = 128; st > 0; st >>= 1) {
        if (tid < st) red[tid] += red[tid + st];
        __syncthreads();
    }
    if (tid == 0) scores[row] = dsigm(red[0]);
}

__global__ void pool_scan_k(const float* scores, const float* summ,
                            float* pool, float* meta) {
    __shared__ float spool[50][SUMD];
    __shared__ float sprio[50];
    __shared__ int swidx, swdo, scount;
    int tid = threadIdx.x;                              // 64 threads
    for (int j = tid; j < 50 * SUMD; j += 64) (&spool[0][0])[j] = 0.f;
    if (tid < 50) sprio[tid] = 0.f;
    if (tid == 0) scount = 0;
    __syncthreads();
    float ssum = 0.f;
    for (int t = 0; t < BT; ++t) {
        if (tid == 0) {
            float sc = scores[t];
            ssum += sc;
            int minI = 0; float minV = sprio[0];
            for (int j = 1; j < 50; ++j)
                if (sprio[j] < minV) { minV = sprio[j]; minI = j; }
            bool valid = sc > 0.5f;                     // TAU1
            bool notfull = scount < 50;
            int idx = notfull ? scount : minI;
            int write = (valid && (notfull || sc > minV)) ? 1 : 0;
            swidx = idx; swdo = write;
            if (write) sprio[idx] = sc;
            if (valid && notfull) scount++;
        }
        __syncthreads();
        if (swdo) spool[swidx][tid] = summ[(long)t * SUMD + tid];
        __syncthreads();
    }
    for (int r = 0; r < 50; ++r) pool[r * SUMD + tid] = spool[r][tid];
    for (int r = 50; r < 64; ++r) pool[r * SUMD + tid] = 0.f;  // pad rows for WMMA
    if (tid == 0) { meta[0] = ssum / (float)BT; meta[1] = (float)scount; }
}

__global__ void retrieve_k(const float* q, const float* kk, const float* vv,
                           const float* meta, float* ret) {
    __shared__ float lg[50];
    __shared__ float sinv;
    long t = blockIdx.x; int tid = threadIdx.x;         // 128 threads
    int count = (int)meta[1];
    if (tid < 50) {
        float d = 0.f;
        for (int j = 0; j < SUMD; ++j) d += q[t * SUMD + j] * kk[tid * SUMD + j];
        d *= 0.25f;                                     // / sqrt(RET_HEAD_DIM=16)
        lg[tid] = (tid < count) ? d : -1e9f;
    }
    __syncthreads();
    if (tid == 0) {
        float mx = lg[0];
        for (int j = 1; j < 50; ++j) mx = fmaxf(mx, lg[j]);
        float s = 0.f;
        for (int j = 0; j < 50; ++j) { lg[j] = __expf(lg[j] - mx); s += lg[j]; }
        sinv = 1.f / s;
    }
    __syncthreads();
    float inv = sinv;
    for (int d = tid; d < DMODEL; d += 128) {
        float acc = 0.f;
        for (int p = 0; p < 50; ++p) acc += lg[p] * vv[p * DMODEL + d];
        ret[t * DMODEL + d] = acc * inv;
    }
}

__global__ void final_k(const float* x, const float* y, const float* gate,
                        const float* ret, const float* meta, float* out) {
    long i = (long)blockIdx.x * 256 + threadIdx.x;
    if (i >= (long)BT * DMODEL) return;
    float cond = (meta[0] > 0.3f && meta[1] > 0.f) ? 1.f : 0.f;   // TAU2 & count>0
    out[i] = x[i] + y[i] + cond * gate[i] * ret[i];
}

// ---------------- host side ----------------
template <class FA, class FB, class FE>
static void launch_gemm(FA fa, FB fb, FE fe, int M, int N, int K, int batch,
                        hipStream_t s) {
    dim3 g(N / 32, (M + 127) / 128, batch);   // block tile 128x32 (4 waves x 32 rows)
    gemm_k<FA, FB, FE><<<g, 128, 0, s>>>(fa, fb, fe, M, N, K);
}

extern "C" void kernel_launch(void* const* d_in, const int* in_sizes, int n_in,
                              void* d_out, int out_size, void* d_ws, size_t ws_size,
                              hipStream_t stream) {
    const float* x          = (const float*)d_in[0];
    const float* norm_w     = (const float*)d_in[1];
    const float* in_proj_w  = (const float*)d_in[2];
    const float* conv_w     = (const float*)d_in[3];
    const float* conv_b     = (const float*)d_in[4];
    const float* dt_bias    = (const float*)d_in[5];
    const float* A_log      = (const float*)d_in[6];
    const float* Dv         = (const float*)d_in[7];
    const float* ssm_norm_w = (const float*)d_in[8];
    const float* out_proj_w = (const float*)d_in[9];
    const float* scorer_w1  = (const float*)d_in[10];
    const float* scorer_w2  = (const float*)d_in[11];
    const float* summ_w     = (const float*)d_in[12];
    const float* q_w        = (const float*)d_in[13];
    const float* k_w        = (const float*)d_in[14];
    const float* v_w        = (const float*)d_in[15];
    const float* gate_w     = (const float*)d_in[16];
    float* out = (float*)d_out;

    float* w = (float*)d_ws;
    size_t off = 0;
    auto alloc = [&](size_t n) { float* p = w + off; off += n; return p; };
    float* XN   = alloc((size_t)BT * DMODEL);
    float* ZX   = alloc((size_t)BT * DPROJ);
    float* XBC  = alloc((size_t)BT * CONVD);
    float* XD   = alloc((size_t)BT * DINNER);
    float* DA   = alloc((size_t)BT * NH);
    float* CS   = alloc((size_t)ZB * CHUNKL);
    float* CBB  = alloc((size_t)BSZ * NCHUNK * CHUNKL * CHUNKL);
    float* YS   = alloc((size_t)BT * DINNER);
    float* ST   = alloc((size_t)ZB * HD * DSTATE);
    float* PV   = alloc((size_t)ZB * HD * DSTATE);
    float* Y2   = alloc((size_t)BT * DINNER);
    float* YB   = alloc((size_t)BT * DMODEL);
    float* HID  = alloc((size_t)BT * SCHID);
    float* SC   = alloc((size_t)BT);
    float* SUMM = alloc((size_t)BT * SUMD);
    float* QB   = alloc((size_t)BT * SUMD);
    float* POOL = alloc((size_t)64 * SUMD);
    float* META = alloc(8);
    float* KKB  = alloc((size_t)64 * SUMD);
    float* VVB  = alloc((size_t)64 * DMODEL);
    float* RET  = alloc((size_t)BT * DMODEL);
    float* GATE = alloc((size_t)BT * DMODEL);
    (void)ws_size; (void)in_sizes; (void)n_in; (void)out_size;

    const float* Bm = XBC + DINNER;            // cols [2048,2176), ld CONVD
    const float* Cm = XBC + DINNER + DSTATE;   // cols [2176,2304), ld CONVD

    // 1. pre-norm
    rmsnorm1_k<<<BT, 256, 0, stream>>>(x, norm_w, XN);
    // 2. in_proj: (2048x1024)x(1024x4384)
    launch_gemm(LdA{XN, DMODEL, 0, 0}, LdB{in_proj_w, DPROJ, 0, 0},
                EpStore{ZX, DPROJ, 0}, BT, DPROJ, DMODEL, 1, stream);
    // 3. causal conv1d + SiLU
    conv_k<<<((long)BT * CONVD + 255) / 256, 256, 0, stream>>>(ZX, conv_w, conv_b, XBC);
    // 4. dt = softplus(...), dA = dt*(-exp(A_log)), xd = xs*dt
    dt_k<<<(BT * NH + 255) / 256, 256, 0, stream>>>(ZX, dt_bias, A_log, XBC, XD, DA);
    // 5. per-chunk per-head cumsum of dA
    cumsum_k<<<BSZ * NCHUNK, NH, 0, stream>>>(DA, CS);
    // 6. CB[b,c] = Cm_chunk (256x128) @ Bm_chunk^T (128x256)
    launch_gemm(LdA{Cm, CONVD, (long)CHUNKL * CONVD, 0},
                LdBT{Bm, CONVD, (long)CHUNKL * CONVD, 0},
                EpStore{CBB, CHUNKL, (long)CHUNKL * CHUNKL},
                CHUNKL, CHUNKL, DSTATE, BSZ * NCHUNK, stream);
    // 7. y_diag[b,c,h] = (CB .* L_h) @ xd_h   (mask+decay folded into A loader)
    launch_gemm(LdSegA{CBB, CS}, LdXdB{XD}, EpYdiag{YS},
                CHUNKL, HD, CHUNKL, ZB, stream);
    // 8. states[b,c,h] = (xd^T .* decay) @ Bm   (64x128, K=256)
    launch_gemm(LdDecayA{XD, CS}, LdB{Bm, CONVD, (long)CHUNKL * CONVD, 5},
                EpStore{ST, DSTATE, (long)HD * DSTATE},
                HD, DSTATE, CHUNKL, ZB, stream);
    // 9. inter-chunk recurrence (c=4, <=3 terms each)
    prev_k<<<(BSZ * NCHUNK * NH * HD * DSTATE + 255) / 256, 256, 0, stream>>>(ST, CS, PV);
    // 10. y_off: Cm @ prev^T, scaled exp(cs[l]), += D*xs, accumulated into YS
    launch_gemm(LdA{Cm, CONVD, (long)CHUNKL * CONVD, 5},
                LdBT{PV, DSTATE, (long)HD * DSTATE, 0},
                EpYoff{YS, CS, XBC, Dv},
                CHUNKL, HD, DSTATE, ZB, stream);
    // 11. gated RMSNorm: rmsnorm(y * silu(z))
    gatednorm_k<<<BT, 256, 0, stream>>>(YS, ZX, ssm_norm_w, Y2);
    // 12. out_proj
    launch_gemm(LdA{Y2, DINNER, 0, 0}, LdB{out_proj_w, DMODEL, 0, 0},
                EpStore{YB, DMODEL, 0}, BT, DMODEL, DINNER, 1, stream);
    // 13-14. scorer
    launch_gemm(LdA{YB, DMODEL, 0, 0}, LdB{scorer_w1, SCHID, 0, 0},
                EpRelu{HID, SCHID}, BT, SCHID, DMODEL, 1, stream);
    scorer2_k<<<BT, 256, 0, stream>>>(HID, scorer_w2, SC);
    // 15-16. summaries and queries
    launch_gemm(LdA{YB, DMODEL, 0, 0}, LdB{summ_w, SUMD, 0, 0},
                EpStore{SUMM, SUMD, 0}, BT, SUMD, DMODEL, 1, stream);
    launch_gemm(LdA{YB, DMODEL, 0, 0}, LdB{q_w, SUMD, 0, 0},
                EpStore{QB, SUMD, 0}, BT, SUMD, DMODEL, 1, stream);
    // 17. serial pool scan (2048 steps)
    pool_scan_k<<<1, 64, 0, stream>>>(SC, SUMM, POOL, META);
    // 18-19. pool projections (rows padded to 64 for WMMA; extras masked later)
    launch_gemm(LdA{POOL, SUMD, 0, 0}, LdB{k_w, SUMD, 0, 0},
                EpStore{KKB, SUMD, 0}, 64, SUMD, SUMD, 1, stream);
    launch_gemm(LdA{POOL, SUMD, 0, 0}, LdB{v_w, DMODEL, 0, 0},
                EpStore{VVB, DMODEL, 0}, 64, DMODEL, SUMD, 1, stream);
    // 20. retrieval attention over <=50 slots
    retrieve_k<<<BT, 128, 0, stream>>>(QB, KKB, VVB, META, RET);
    // 21-22. gate = sigmoid([y, retrieved] @ gate_w), split-K two-pass
    launch_gemm(LdA{YB, DMODEL, 0, 0}, LdB{gate_w, DMODEL, 0, 0},
                EpStore{GATE, DMODEL, 0}, BT, DMODEL, DMODEL, 1, stream);
    launch_gemm(LdA{RET, DMODEL, 0, 0},
                LdB{gate_w + (size_t)DMODEL * DMODEL, DMODEL, 0, 0},
                EpAddSig{GATE, DMODEL}, BT, DMODEL, DMODEL, 1, stream);
    // 23. residual + conditional gated retrieval
    final_k<<<((long)BT * DMODEL + 255) / 256, 256, 0, stream>>>(
        x, YB, GATE, RET, META, out);
}
